// dgcnn_encoder_7481833029678
// MI455X (gfx1250) — matrix operand order repaired and checked by reading, not compiled, not run on hardware
//
#include <hip/hip_runtime.h>
#include <stdint.h>

typedef __attribute__((ext_vector_type(16))) __bf16 v16bf;
typedef __attribute__((ext_vector_type(8)))  float  v8f;

union BF16x16 { v16bf v; uint4 q[2]; };

#define NPTS 2048
#define NBATCH 8
#define KNN 20
#define BN_EPS 1e-5f

// float atomic max/min for non-negative floats (post-ReLU): uint bit order == float order
static __device__ __forceinline__ void atomicMaxF(float* a, float v) {
    atomicMax((unsigned int*)a, __float_as_uint(v));
}
static __device__ __forceinline__ void atomicMinF(float* a, float v) {
    atomicMin((unsigned int*)a, __float_as_uint(v));
}

// ---------------------------------------------------------------------------
// Prep: x [B,3,N] -> x0f [B,N,3] f32, xb0 [B,N,32] bf16 (zero padded), sq [B,N]
// ---------------------------------------------------------------------------
__global__ void k_prep0(const float* __restrict__ x, float* __restrict__ x0f,
                        __bf16* __restrict__ xb0, float* __restrict__ sq) {
    int gid = blockIdx.x * blockDim.x + threadIdx.x;
    if (gid >= NBATCH * NPTS) return;
    int b = gid >> 11, n = gid & 2047;
    float s = 0.f;
#pragma unroll
    for (int c = 0; c < 3; ++c) {
        float v = x[(b * 3 + c) * NPTS + n];
        x0f[gid * 3 + c] = v;
        xb0[gid * 32 + c] = (__bf16)v;
        s += v * v;
    }
#pragma unroll
    for (int c = 3; c < 32; ++c) xb0[gid * 32 + c] = (__bf16)0.f;
    sq[gid] = s;
}

// ---------------------------------------------------------------------------
// Repack W [K2C, D] f32 -> bf16 B-fragment layout:
//   wf[(((ct*KSTEPS + ks)*2 + half)*16 + colq)*16 + j] = W[ks*32 + half*16 + j][ct*16 + colq]
// ---------------------------------------------------------------------------
__global__ void k_repack(const float* __restrict__ W, __bf16* __restrict__ wf,
                         int twoC, int D, int KSTEPS, int total) {
    int gid = blockIdx.x * blockDim.x + threadIdx.x;
    if (gid >= total) return;
    int j  = gid & 15;
    int lq = (gid >> 4) & 15;
    int hf = (gid >> 8) & 1;
    int r  = gid >> 9;
    int ks = r % KSTEPS, ct = r / KSTEPS;
    int k   = ks * 32 + hf * 16 + j;
    int col = ct * 16 + lq;
    float v = (k < twoC) ? W[k * D + col] : 0.f;
    wf[gid] = (__bf16)v;
}

// ---------------------------------------------------------------------------
// Gram/distance: dist[n][m] = sq[n] - 2*<x_n,x_m> + sq[m]  (one batch)
// bf16 WMMA; each wave owns a 16-row strip, loops over 128 col tiles with
// double-buffered B fragments so loads overlap the WMMA chain.
// ---------------------------------------------------------------------------
template <int KST>
__global__ void k_gram(const __bf16* __restrict__ xb, const float* __restrict__ sq,
                       float* __restrict__ dist, int Kpad) {
    int tid = threadIdx.x;
    int wid = tid >> 5, lane = tid & 31;
    int hf = lane >> 4, lq = lane & 15;
    int rt = blockIdx.x * 4 + wid;
    int n0 = rt * 16;

    BF16x16 afr[KST];
#pragma unroll
    for (int ks = 0; ks < KST; ++ks) {
        const __bf16* p = xb + (n0 + lq) * Kpad + ks * 32 + hf * 16;
        afr[ks].q[0] = *(const uint4*)p;         // K = base + {0..7} (or {16..23})
        afr[ks].q[1] = *(const uint4*)(p + 8);   // K = base + {8..15} (or {24..31})
    }
    float rs[8];
#pragma unroll
    for (int i = 0; i < 8; ++i) rs[i] = sq[n0 + hf * 8 + i];

    BF16x16 bcur[KST], bnext[KST];
#pragma unroll
    for (int ks = 0; ks < KST; ++ks) {
        const __bf16* p = xb + lq * Kpad + ks * 32 + hf * 16;
        bcur[ks].q[0] = *(const uint4*)p;        // K = base + {0..7}  col = lq
        bcur[ks].q[1] = *(const uint4*)(p + 8);  // K = base + {8..15}
    }
    for (int ct = 0; ct < 128; ++ct) {
        if (ct < 127) {
            int c1 = (ct + 1) * 16;
#pragma unroll
            for (int ks = 0; ks < KST; ++ks) {
                const __bf16* p = xb + (c1 + lq) * Kpad + ks * 32 + hf * 16;
                bnext[ks].q[0] = *(const uint4*)p;
                bnext[ks].q[1] = *(const uint4*)(p + 8);
            }
        }
        v8f acc;
#pragma unroll
        for (int i = 0; i < 8; ++i) acc[i] = 0.f;
#pragma unroll
        for (int ks = 0; ks < KST; ++ks) {
            acc = __builtin_amdgcn_wmma_f32_16x16x32_bf16(
                false, afr[ks].v, false, bcur[ks].v, (short)0, acc, false, false);
        }
        int c0 = ct * 16;
        float cs = sq[c0 + lq];
#pragma unroll
        for (int i = 0; i < 8; ++i)
            dist[(n0 + hf * 8 + i) * NPTS + c0 + lq] = rs[i] - 2.f * acc[i] + cs;
#pragma unroll
        for (int ks = 0; ks < KST; ++ks) {
            bcur[ks].q[0] = bnext[ks].q[0];
            bcur[ks].q[1] = bnext[ks].q[1];
        }
    }
}

// ---------------------------------------------------------------------------
// Top-k (k=20 smallest, self included). One thread per row; dist is L2 resident.
// ---------------------------------------------------------------------------
__global__ void k_topk(const float* __restrict__ dist, int* __restrict__ idxo) {
    int n = blockIdx.x * blockDim.x + threadIdx.x;
    if (n >= NPTS) return;
    float dv[KNN];
    int di[KNN];
#pragma unroll
    for (int t = 0; t < KNN; ++t) { dv[t] = 3.4e38f; di[t] = 0; }
    float worst = 3.4e38f;
    int wpos = 0;
    const float* row = dist + (size_t)n * NPTS;
    for (int m = 0; m < NPTS; ++m) {
        float d = row[m];
        if (d < worst) {
            dv[wpos] = d; di[wpos] = m;
            worst = -3.4e38f;
            for (int t = 0; t < KNN; ++t)
                if (dv[t] > worst) { worst = dv[t]; wpos = t; }
        }
    }
    for (int t = 0; t < KNN; ++t) idxo[n * KNN + t] = di[t];
}

// ---------------------------------------------------------------------------
// Fused EdgeConv: per (batch, 16-point tile) gather edges for each k into LDS,
// WMMA against register-resident weight fragments, keep running max/min over k
// in the accumulator tile, atomically accumulate channel sums for BN stats.
// blockDim = 32 * (D/16); wave `wid` owns column tile wid.
// ---------------------------------------------------------------------------
template <int KSTEPS>
__global__ void k_edgeconv(const float* __restrict__ xl, const int* __restrict__ nidx,
                           const __bf16* __restrict__ wfrag, const float* __restrict__ bias,
                           float* __restrict__ sum, float* __restrict__ sumsq,
                           float* __restrict__ hmax, float* __restrict__ hmin,
                           int C, int D) {
    extern __shared__ __bf16 etile[];  // [16][Ktot]
    const int Ktot = KSTEPS * 32;
    const int twoC = 2 * C;
    int tile = blockIdx.x;
    int b = tile >> 7, nt = tile & 127, n0 = nt * 16;
    int tid = threadIdx.x, wid = tid >> 5, lane = tid & 31;
    int hf = lane >> 4, lq = lane & 15;

    BF16x16 bfr[KSTEPS];
#pragma unroll
    for (int ks = 0; ks < KSTEPS; ++ks) {
        const __bf16* p = wfrag + ((((wid * KSTEPS + ks) * 2 + hf) * 16 + lq) << 4);
        bfr[ks].q[0] = *(const uint4*)p;
        bfr[ks].q[1] = *(const uint4*)(p + 8);
    }
    float bval = bias[wid * 16 + lq];
    v8f vmax, vmin;
#pragma unroll
    for (int i = 0; i < 8; ++i) { vmax[i] = -3.4e38f; vmin[i] = 3.4e38f; }
    float s1 = 0.f, s2 = 0.f;

    for (int k = 0; k < KNN; ++k) {
        __syncthreads();  // protect previous iteration's LDS reads
        for (int u = tid; u < 16 * Ktot; u += blockDim.x) {
            int i = u / Ktot, c = u % Ktot;
            int n = n0 + i;
            float v = 0.f;
            if (c < C) {
                v = xl[(b * NPTS + n) * C + c];
            } else if (c < twoC) {
                int j = nidx[(b * NPTS + n) * KNN + k];
                int cc = c - C;
                v = xl[(b * NPTS + j) * C + cc] - xl[(b * NPTS + n) * C + cc];
            }
            etile[u] = (__bf16)v;
        }
        __syncthreads();

        v8f acc;
#pragma unroll
        for (int i = 0; i < 8; ++i) acc[i] = 0.f;
#pragma unroll
        for (int ks = 0; ks < KSTEPS; ++ks) {
            BF16x16 af;
            const __bf16* p = etile + lq * Ktot + ks * 32 + hf * 16;
            af.q[0] = *(const uint4*)p;
            af.q[1] = *(const uint4*)(p + 8);
            acc = __builtin_amdgcn_wmma_f32_16x16x32_bf16(
                false, af.v, false, bfr[ks].v, (short)0, acc, false, false);
        }
#pragma unroll
        for (int i = 0; i < 8; ++i) {
            float v = fmaxf(acc[i] + bval, 0.f);
            s1 += v; s2 += v * v;
            vmax[i] = fmaxf(vmax[i], v);
            vmin[i] = fminf(vmin[i], v);
        }
    }
    int col = wid * 16 + lq;
#pragma unroll
    for (int i = 0; i < 8; ++i) {
        int row = n0 + hf * 8 + i;
        hmax[(b * NPTS + row) * D + col] = vmax[i];
        hmin[(b * NPTS + row) * D + col] = vmin[i];
    }
    atomicAdd(&sum[col], s1);
    atomicAdd(&sumsq[col], s2);
}

// ---------------------------------------------------------------------------
// BN finalize + apply (BN commutes with max via slope sign). Also emits bf16
// copy (for next kNN) and per-point squared norm. blockDim = D.
// ---------------------------------------------------------------------------
__global__ void k_bn_apply(const float* __restrict__ hmax, const float* __restrict__ hmin,
                           const float* __restrict__ sum, const float* __restrict__ sumsq,
                           const float* __restrict__ g, const float* __restrict__ be,
                           float invCnt, float* __restrict__ xout,
                           __bf16* __restrict__ xbf, float* __restrict__ sqout, int D) {
    __shared__ float red[128];
    int bn = blockIdx.x;
    int c = threadIdx.x;
    float mean = sum[c] * invCnt;
    float var = sumsq[c] * invCnt - mean * mean;
    float gc = g[c];
    float sc = gc * rsqrtf(var + BN_EPS);
    float base = be[c] - mean * sc;
    float h = (gc >= 0.f) ? hmax[bn * D + c] : hmin[bn * D + c];
    float v = h * sc + base;
    xout[bn * D + c] = v;
    if (xbf) xbf[bn * 64 + c] = (__bf16)v;
    red[c] = v * v;
    __syncthreads();
    for (int s = blockDim.x >> 1; s > 0; s >>= 1) {
        if (c < s) red[c] += red[c + s];
        __syncthreads();
    }
    if (c == 0 && sqout) sqout[bn] = red[0];
}

// ---------------------------------------------------------------------------
// Final FC: relu(concat(x1..x4) @ W5 + b5), channel stats + per-(b,channel)
// max/min over points via atomics. 8 waves, each loops 8 of 64 column tiles
// with a rotating double-buffered B fragment so loads overlap WMMA.
// ---------------------------------------------------------------------------
__global__ void k_fc5(const float* __restrict__ x1, const float* __restrict__ x2,
                      const float* __restrict__ x3, const float* __restrict__ x4,
                      const __bf16* __restrict__ wfrag, const float* __restrict__ bias,
                      float* __restrict__ sum, float* __restrict__ sumsq,
                      float* __restrict__ gmax, float* __restrict__ gmin) {
    __shared__ __bf16 etile[16 * 320];
    int tile = blockIdx.x;
    int b = tile >> 7, nt = tile & 127, n0 = nt * 16;
    int tid = threadIdx.x, wid = tid >> 5, lane = tid & 31;
    int hf = lane >> 4, lq = lane & 15;

    for (int u = tid; u < 16 * 320; u += blockDim.x) {
        int i = u / 320, c = u % 320;
        int n = n0 + i;
        float v;
        if (c < 64)       v = x1[(b * NPTS + n) * 64 + c];
        else if (c < 128) v = x2[(b * NPTS + n) * 64 + (c - 64)];
        else if (c < 192) v = x3[(b * NPTS + n) * 64 + (c - 128)];
        else              v = x4[(b * NPTS + n) * 128 + (c - 192)];
        etile[u] = (__bf16)v;
    }
    __syncthreads();

    BF16x16 afr[10];
#pragma unroll
    for (int ks = 0; ks < 10; ++ks) {
        const __bf16* p = etile + lq * 320 + ks * 32 + hf * 16;
        afr[ks].q[0] = *(const uint4*)p;
        afr[ks].q[1] = *(const uint4*)(p + 8);
    }
    const __bf16* wbase = wfrag + ((hf * 16 + lq) << 4);
    for (int ct = wid; ct < 64; ct += 8) {
        v8f acc;
#pragma unroll
        for (int i = 0; i < 8; ++i) acc[i] = 0.f;
        BF16x16 bcur, bnext;
        {
            const __bf16* p = wbase + (size_t)(ct * 10) * 512;
            bcur.q[0] = *(const uint4*)p;
            bcur.q[1] = *(const uint4*)(p + 8);
        }
#pragma unroll
        for (int ks = 0; ks < 10; ++ks) {
            if (ks < 9) {
                const __bf16* p = wbase + (size_t)(ct * 10 + ks + 1) * 512;
                bnext.q[0] = *(const uint4*)p;
                bnext.q[1] = *(const uint4*)(p + 8);
            }
            acc = __builtin_amdgcn_wmma_f32_16x16x32_bf16(
                false, afr[ks].v, false, bcur.v, (short)0, acc, false, false);
            bcur.q[0] = bnext.q[0];
            bcur.q[1] = bnext.q[1];
        }
        int col = ct * 16 + lq;
        float bval = bias[col];
        float s1 = 0.f, s2 = 0.f, mx = -3.4e38f, mn = 3.4e38f;
#pragma unroll
        for (int i = 0; i < 8; ++i) {
            float v = fmaxf(acc[i] + bval, 0.f);
            s1 += v; s2 += v * v;
            mx = fmaxf(mx, v); mn = fminf(mn, v);
        }
        atomicAdd(&sum[col], s1);
        atomicAdd(&sumsq[col], s2);
        atomicMaxF(&gmax[b * 1024 + col], mx);
        atomicMinF(&gmin[b * 1024 + col], mn);
    }
}

__global__ void k_out5(const float* __restrict__ gmax, const float* __restrict__ gmin,
                       const float* __restrict__ sum, const float* __restrict__ sumsq,
                       const float* __restrict__ g, const float* __restrict__ be,
                       float* __restrict__ out) {
    int gid = blockIdx.x * blockDim.x + threadIdx.x;
    if (gid >= NBATCH * 1024) return;
    int c = gid & 1023;
    const float invCnt = 1.f / 16384.f;
    float mean = sum[c] * invCnt;
    float var = sumsq[c] * invCnt - mean * mean;
    float gc = g[c];
    float sc = gc * rsqrtf(var + BN_EPS);
    float base = be[c] - mean * sc;
    float h = (gc >= 0.f) ? gmax[gid] : gmin[gid];
    out[gid] = h * sc + base;
}

__global__ void k_fill(float* __restrict__ p, int n, float v) {
    int gid = blockIdx.x * blockDim.x + threadIdx.x;
    if (gid < n) p[gid] = v;
}

// ---------------------------------------------------------------------------
extern "C" void kernel_launch(void* const* d_in, const int* in_sizes, int n_in,
                              void* d_out, int out_size, void* d_ws, size_t ws_size,
                              hipStream_t stream) {
    (void)in_sizes; (void)n_in; (void)out_size; (void)ws_size;
    const float* x   = (const float*)d_in[0];
    const float* W1  = (const float*)d_in[1];
    const float* b1  = (const float*)d_in[2];
    const float* g1  = (const float*)d_in[3];
    const float* be1 = (const float*)d_in[4];
    const float* W2  = (const float*)d_in[5];
    const float* b2  = (const float*)d_in[6];
    const float* g2  = (const float*)d_in[7];
    const float* be2 = (const float*)d_in[8];
    const float* W3  = (const float*)d_in[9];
    const float* b3  = (const float*)d_in[10];
    const float* g3  = (const float*)d_in[11];
    const float* be3 = (const float*)d_in[12];
    const float* W4  = (const float*)d_in[13];
    const float* b4  = (const float*)d_in[14];
    const float* g4  = (const float*)d_in[15];
    const float* be4 = (const float*)d_in[16];
    const float* W5  = (const float*)d_in[17];
    const float* b5  = (const float*)d_in[18];
    const float* g5  = (const float*)d_in[19];
    const float* be5 = (const float*)d_in[20];

    char* ws = (char*)d_ws;
    size_t off = 0;
    auto carve = [&](size_t bytes) -> char* {
        char* p = ws + off;
        off += (bytes + 255) & ~(size_t)255;
        return p;
    };
    const int BN_TOT = NBATCH * NPTS;  // 16384
    float*  x0f  = (float*)carve((size_t)BN_TOT * 3 * 4);
    __bf16* xb0  = (__bf16*)carve((size_t)BN_TOT * 32 * 2);
    __bf16* xbL  = (__bf16*)carve((size_t)BN_TOT * 64 * 2);
    float*  sq   = (float*)carve((size_t)BN_TOT * 4);
    float*  x1   = (float*)carve((size_t)BN_TOT * 64 * 4);
    float*  x2   = (float*)carve((size_t)BN_TOT * 64 * 4);
    float*  x3   = (float*)carve((size_t)BN_TOT * 64 * 4);
    float*  x4   = (float*)carve((size_t)BN_TOT * 128 * 4);
    float*  dist = (float*)carve((size_t)NPTS * NPTS * 4);
    int*    idx  = (int*)carve((size_t)BN_TOT * KNN * 4);
    float*  hmax = (float*)carve((size_t)BN_TOT * 128 * 4);
    float*  hmin = (float*)carve((size_t)BN_TOT * 128 * 4);
    float*  sums = (float*)carve(2048 * 4);
    float*  sum  = sums;
    float*  sumsq = sums + 1024;
    float*  gmax = (float*)carve(8192 * 4);
    float*  gmin = (float*)carve(8192 * 4);
    __bf16* wf1  = (__bf16*)carve(32 * 64 * 2);
    __bf16* wf2  = (__bf16*)carve(128 * 64 * 2);
    __bf16* wf3  = (__bf16*)carve(128 * 64 * 2);
    __bf16* wf4  = (__bf16*)carve(128 * 128 * 2);
    __bf16* wf5  = (__bf16*)carve(320 * 1024 * 2);

    const float invEdgeCnt = 1.f / (float)(NBATCH * NPTS * KNN);  // 1/327680

    // Prep + weight repacks
    k_prep0<<<64, 256, 0, stream>>>(x, x0f, xb0, sq);
    k_repack<<<(32 * 64 + 255) / 256, 256, 0, stream>>>(W1, wf1, 6, 64, 1, 32 * 64);
    k_repack<<<(128 * 64 + 255) / 256, 256, 0, stream>>>(W2, wf2, 128, 64, 4, 128 * 64);
    k_repack<<<(128 * 64 + 255) / 256, 256, 0, stream>>>(W3, wf3, 128, 64, 4, 128 * 64);
    k_repack<<<(128 * 128 + 255) / 256, 256, 0, stream>>>(W4, wf4, 128, 128, 4, 128 * 128);
    k_repack<<<(320 * 1024 + 255) / 256, 256, 0, stream>>>(W5, wf5, 320, 1024, 10, 320 * 1024);

    // ---- Layer 1 (C=3 -> D=64, conv K padded 6->32) ----
    for (int b = 0; b < NBATCH; ++b) {
        k_gram<1><<<32, 128, 0, stream>>>(xb0 + (size_t)b * NPTS * 32, sq + b * NPTS, dist, 32);
        k_topk<<<8, 256, 0, stream>>>(dist, idx + (size_t)b * NPTS * KNN);
    }
    k_fill<<<8, 256, 0, stream>>>(sums, 2048, 0.f);
    k_edgeconv<1><<<1024, 128, 16 * 32 * 2, stream>>>(x0f, idx, wf1, b1, sum, sumsq, hmax, hmin, 3, 64);
    k_bn_apply<<<BN_TOT, 64, 0, stream>>>(hmax, hmin, sum, sumsq, g1, be1, invEdgeCnt, x1, xbL, sq, 64);

    // ---- Layer 2 (C=64 -> D=64) ----
    for (int b = 0; b < NBATCH; ++b) {
        k_gram<2><<<32, 128, 0, stream>>>(xbL + (size_t)b * NPTS * 64, sq + b * NPTS, dist, 64);
        k_topk<<<8, 256, 0, stream>>>(dist, idx + (size_t)b * NPTS * KNN);
    }
    k_fill<<<8, 256, 0, stream>>>(sums, 2048, 0.f);
    k_edgeconv<4><<<1024, 128, 16 * 128 * 2, stream>>>(x1, idx, wf2, b2, sum, sumsq, hmax, hmin, 64, 64);
    k_bn_apply<<<BN_TOT, 64, 0, stream>>>(hmax, hmin, sum, sumsq, g2, be2, invEdgeCnt, x2, xbL, sq, 64);

    // ---- Layer 3 (C=64 -> D=64) ----
    for (int b = 0; b < NBATCH; ++b) {
        k_gram<2><<<32, 128, 0, stream>>>(xbL + (size_t)b * NPTS * 64, sq + b * NPTS, dist, 64);
        k_topk<<<8, 256, 0, stream>>>(dist, idx + (size_t)b * NPTS * KNN);
    }
    k_fill<<<8, 256, 0, stream>>>(sums, 2048, 0.f);
    k_edgeconv<4><<<1024, 128, 16 * 128 * 2, stream>>>(x2, idx, wf3, b3, sum, sumsq, hmax, hmin, 64, 64);
    k_bn_apply<<<BN_TOT, 64, 0, stream>>>(hmax, hmin, sum, sumsq, g3, be3, invEdgeCnt, x3, xbL, sq, 64);

    // ---- Layer 4 (C=64 -> D=128) ----
    for (int b = 0; b < NBATCH; ++b) {
        k_gram<2><<<32, 128, 0, stream>>>(xbL + (size_t)b * NPTS * 64, sq + b * NPTS, dist, 64);
        k_topk<<<8, 256, 0, stream>>>(dist, idx + (size_t)b * NPTS * KNN);
    }
    k_fill<<<8, 256, 0, stream>>>(sums, 2048, 0.f);
    k_edgeconv<4><<<1024, 256, 16 * 128 * 2, stream>>>(x3, idx, wf4, b4, sum, sumsq, hmax, hmin, 64, 128);
    k_bn_apply<<<BN_TOT, 128, 0, stream>>>(hmax, hmin, sum, sumsq, g4, be4, invEdgeCnt, x4,
                                           (__bf16*)nullptr, (float*)nullptr, 128);

    // ---- Layer 5 (320 -> 1024, BN, max over points) ----
    k_fill<<<8, 256, 0, stream>>>(sums, 2048, 0.f);
    k_fill<<<32, 256, 0, stream>>>(gmax, 8192, 0.f);
    k_fill<<<32, 256, 0, stream>>>(gmin, 8192, 3.4e38f);
    k_fc5<<<1024, 256, 0, stream>>>(x1, x2, x3, x4, wf5, b5, sum, sumsq, gmax, gmin);
    k_out5<<<32, 256, 0, stream>>>(gmax, gmin, sum, sumsq, g5, be5, (float*)d_out);
}
